// GraphEncoder_18769007083890
// MI455X (gfx1250) — matrix operand order repaired and checked by reading, not compiled, not run on hardware
//
#include <hip/hip_runtime.h>
#include <cstdint>
#include <cstddef>

#define DEV __device__ __forceinline__

typedef __attribute__((ext_vector_type(16))) __bf16 v16bf;
typedef __attribute__((ext_vector_type(8)))  float  v8f;
typedef __attribute__((ext_vector_type(4)))  int    v4i;
typedef __attribute__((address_space(1))) v4i gv4i;   // global int4
typedef __attribute__((address_space(3))) v4i lv4i;   // LDS int4

union FragU { uint4 u[2]; v16bf v; };

DEV unsigned short f2bf(float f) {
  unsigned u = __float_as_uint(f);
  u += 0x7fffu + ((u >> 16) & 1u);          // round-to-nearest-even
  return (unsigned short)(u >> 16);
}

// A-fragment (16x32 bf16): lane holds row m=lane&15; halves at k=(lane>>4)*8+{0..7} and +16.
DEV v16bf fragA(const unsigned short* p, int pitch, int row, int kbase, int lane) {
  const unsigned short* r = p + row * pitch + kbase + ((lane >> 4) << 3);
  FragU f; f.u[0] = *(const uint4*)r; f.u[1] = *(const uint4*)(r + 16);
  return f.v;
}
// B-fragment (32x16 bf16): lane holds col n=lane&15; halves at k=(lane>>4)*8+{0..7} and +16.
DEV v16bf fragB(const unsigned short* p, int pitch, int row, int kbase, int lane) {
  const unsigned short* r = p + row * pitch + kbase + ((lane >> 4) << 3);
  FragU f; f.u[0] = *(const uint4*)r; f.u[1] = *(const uint4*)(r + 16);
  return f.v;
}
DEV v8f wmma_bf16(v16bf a, v16bf b, v8f c) {
  return __builtin_amdgcn_wmma_f32_16x16x32_bf16(false, a, false, b, (short)0, c, false, false);
}

// Async global->LDS 16B copy (GLOBAL_LOAD_ASYNC_TO_LDS_B128, tracked by ASYNCcnt).
DEV void cp_async_b128(const unsigned short* g, unsigned short* l) {
  __builtin_amdgcn_global_load_async_to_lds_b128((gv4i*)g, (lv4i*)l, 0, 0);
}

// ---------------------------------------------------------------------------
// GEMM: out[N,M] = A(bf16 [N,K]) @ Bt^T (Bt = bf16 [M,K] pre-transposed) + bias
// MODE 0: f32 out;  MODE 1: f32 out + residual;  MODE 2: bf16 relu(out)
// block tile 128x128, 8 waves (4x2), per wave 2x4 16x16 tiles, K step 32.
// Double-buffered async global->LDS staging: copy of tile i+1 overlaps WMMAs
// of tile i; each wave issues 4 b128 async ops/tile so s_wait_asynccnt(4)
// means "tile i landed, tile i+1 still in flight".
// ---------------------------------------------------------------------------
template<int MODE>
__global__ __launch_bounds__(256) void k_gemm(
    const unsigned short* __restrict__ A, const unsigned short* __restrict__ Bt,
    const float* __restrict__ bias, const float* __restrict__ res,
    float* __restrict__ outF, unsigned short* __restrict__ outB, int K, int M)
{
  __shared__ unsigned short As[2][128 * 40];   // pitch 40 halves: conflict-free b128 reads
  __shared__ unsigned short Bs[2][128 * 40];
  const int t = threadIdx.x, lane = t & 31, wave = t >> 5;
  const int ln15 = lane & 15, lh = lane >> 4;
  const int rowBase = blockIdx.x * 128, colBase = blockIdx.y * 128;
  const int wRow = (wave >> 1) * 32, wCol = (wave & 1) * 64;

  // per-thread copy chunks: chunk c -> row c>>2, half-offset (c&3)*8 (16B each)
  const int r0 = t >> 2,         o0 = (t & 3) << 3;
  const int r1 = (t + 256) >> 2, o1 = (t & 3) << 3;   // (t+256)&3 == t&3
  const unsigned short* gA0 = A  + (size_t)(rowBase + r0) * K + o0;
  const unsigned short* gA1 = A  + (size_t)(rowBase + r1) * K + o1;
  const unsigned short* gB0 = Bt + (size_t)(colBase + r0) * K + o0;
  const unsigned short* gB1 = Bt + (size_t)(colBase + r1) * K + o1;
  const int l0 = r0 * 40 + o0, l1 = r1 * 40 + o1;

  v8f acc[2][4] = {};
  const int nIter = K >> 5;

  // prologue: stage tile 0
  cp_async_b128(gA0, &As[0][l0]);
  cp_async_b128(gB0, &Bs[0][l0]);
  cp_async_b128(gA1, &As[0][l1]);
  cp_async_b128(gB1, &Bs[0][l1]);

  for (int i = 0; i < nIter; ++i) {
    const int cur = i & 1;
    if (i + 1 < nIter) {
      const int k0 = (i + 1) << 5, nxt = cur ^ 1;
      cp_async_b128(gA0 + k0, &As[nxt][l0]);
      cp_async_b128(gB0 + k0, &Bs[nxt][l0]);
      cp_async_b128(gA1 + k0, &As[nxt][l1]);
      cp_async_b128(gB1 + k0, &Bs[nxt][l1]);
      __builtin_amdgcn_s_wait_asynccnt(4);   // tile i complete (in-order), i+1 in flight
    } else {
      __builtin_amdgcn_s_wait_asynccnt(0);
    }
    __syncthreads();                          // everyone's tile-i data visible
    v16bf af[2], bfr[4];
    #pragma unroll
    for (int rs = 0; rs < 2; ++rs) af[rs]  = fragA(As[cur], 40, wRow + rs * 16 + ln15, 0, lane);
    #pragma unroll
    for (int cs = 0; cs < 4; ++cs) bfr[cs] = fragB(Bs[cur], 40, wCol + cs * 16 + ln15, 0, lane);
    #pragma unroll
    for (int rs = 0; rs < 2; ++rs)
      #pragma unroll
      for (int cs = 0; cs < 4; ++cs)
        acc[rs][cs] = wmma_bf16(af[rs], bfr[cs], acc[rs][cs]);
    __syncthreads();                          // reads done before tile i+2 overwrites buf[cur]
  }
  #pragma unroll
  for (int rs = 0; rs < 2; ++rs)
    #pragma unroll
    for (int cs = 0; cs < 4; ++cs)
      #pragma unroll
      for (int v = 0; v < 8; ++v) {
        const int gm = rowBase + wRow + rs * 16 + v + lh * 8;
        const int gn = colBase + wCol + cs * 16 + ln15;
        float val = acc[rs][cs][v] + bias[gn];
        if (MODE == 1) val += res[(size_t)gm * M + gn];
        if (MODE == 2) outB[(size_t)gm * M + gn] = f2bf(fmaxf(val, 0.0f));
        else           outF[(size_t)gm * M + gn] = val;
      }
}

// ---------------------------------------------------------------------------
// Fused per-graph multihead attention: 1 block = 1 graph (128 nodes), 8 waves.
// QKV GEMMs from bf16 h (L2-resident), softmax in registers, P@V via LDS.
// attb[node][head*64+dh] = bf16 attention output (pre-Wo).
// ---------------------------------------------------------------------------
__global__ __launch_bounds__(256) void k_attn(
    const unsigned short* __restrict__ hb, const unsigned short* __restrict__ Wqkvt,
    const float* __restrict__ bqkv, unsigned short* __restrict__ attb)
{
  __shared__ unsigned short sQK[128 * 72 * 2];  // Q[128][72] + K[128][72]; P[128][136] overlays
  __shared__ unsigned short sVT[64 * 136];      // V^T [dh][key]
  unsigned short* Q  = sQK;
  unsigned short* Kl = sQK + 128 * 72;
  unsigned short* P  = sQK;
  const int t = threadIdx.x, lane = t & 31, wave = t >> 5;
  const int ln15 = lane & 15, lh = lane >> 4;
  const int nodeBase = blockIdx.x * 128;
  const unsigned short* X = hb + (size_t)nodeBase * 256;
  const float scale = 0.125f;  // 1/sqrt(64)

  for (int hd = 0; hd < 4; ++hd) {
    // ---- QKV projections (wave w owns rows 16w..16w+15) ----
    for (int which = 0; which < 3; ++which) {
      const int wBase = which * 256 + hd * 64;
      v8f acc[4] = {};
      for (int k0 = 0; k0 < 256; k0 += 32) {
        v16bf a = fragA(X, 256, wave * 16 + ln15, k0, lane);
        #pragma unroll
        for (int cs = 0; cs < 4; ++cs)
          acc[cs] = wmma_bf16(a, fragB(Wqkvt, 256, wBase + cs * 16 + ln15, k0, lane), acc[cs]);
      }
      #pragma unroll
      for (int cs = 0; cs < 4; ++cs)
        #pragma unroll
        for (int v = 0; v < 8; ++v) {
          const int mm = wave * 16 + v + lh * 8;   // node row
          const int nn = cs * 16 + ln15;           // dh col
          unsigned short bv = f2bf(acc[cs][v] + bqkv[wBase + nn]);
          if (which == 0)      Q[mm * 72 + nn] = bv;
          else if (which == 1) Kl[mm * 72 + nn] = bv;
          else                 sVT[nn * 136 + mm] = bv;   // store V transposed
        }
    }
    __syncthreads();
    // ---- S = Q @ K^T  (wave strip 16x128) ----
    v8f s[8] = {};
    #pragma unroll
    for (int k0 = 0; k0 < 64; k0 += 32) {
      v16bf a = fragA(Q, 72, wave * 16 + ln15, k0, lane);
      #pragma unroll
      for (int cs = 0; cs < 8; ++cs)
        s[cs] = wmma_bf16(a, fragB(Kl, 72, cs * 16 + ln15, k0, lane), s[cs]);
    }
    // ---- softmax per row (row lives in 16 lanes sharing lane>>4) ----
    float inv[8];
    #pragma unroll
    for (int v = 0; v < 8; ++v) {
      float m = -3.4e38f;
      #pragma unroll
      for (int cs = 0; cs < 8; ++cs) m = fmaxf(m, s[cs][v]);
      m = fmaxf(m, __shfl_xor(m, 1, 32));
      m = fmaxf(m, __shfl_xor(m, 2, 32));
      m = fmaxf(m, __shfl_xor(m, 4, 32));
      m = fmaxf(m, __shfl_xor(m, 8, 32));
      m *= scale;
      float ssum = 0.f;
      #pragma unroll
      for (int cs = 0; cs < 8; ++cs) { float p = __expf(s[cs][v] * scale - m); s[cs][v] = p; ssum += p; }
      ssum += __shfl_xor(ssum, 1, 32);
      ssum += __shfl_xor(ssum, 2, 32);
      ssum += __shfl_xor(ssum, 4, 32);
      ssum += __shfl_xor(ssum, 8, 32);
      inv[v] = 1.0f / ssum;
    }
    __syncthreads();  // all waves done reading Q/K before P overlays them
    #pragma unroll
    for (int cs = 0; cs < 8; ++cs)
      #pragma unroll
      for (int v = 0; v < 8; ++v)
        P[(wave * 16 + v + lh * 8) * 136 + cs * 16 + ln15] = f2bf(s[cs][v] * inv[v]);
    __syncthreads();
    // ---- out = P @ V ----
    v8f o[4] = {};
    #pragma unroll
    for (int k0 = 0; k0 < 128; k0 += 32) {
      v16bf a = fragA(P, 136, wave * 16 + ln15, k0, lane);
      #pragma unroll
      for (int cs = 0; cs < 4; ++cs)
        o[cs] = wmma_bf16(a, fragB(sVT, 136, cs * 16 + ln15, k0, lane), o[cs]);
    }
    #pragma unroll
    for (int cs = 0; cs < 4; ++cs)
      #pragma unroll
      for (int v = 0; v < 8; ++v)
        attb[(size_t)(nodeBase + wave * 16 + v + lh * 8) * 256 + hd * 64 + cs * 16 + ln15]
            = f2bf(o[cs][v]);
    __syncthreads();
  }
}

// ------------------------- small support kernels ---------------------------
__global__ void k_transpose(const float* __restrict__ src, unsigned short* __restrict__ dst,
                            int K, int M) {  // dst[m*K+k] = bf16(src[k*M+m])
  int idx = blockIdx.x * 256 + threadIdx.x;
  if (idx >= K * M) return;
  int m = idx / K, k = idx - m * K;
  dst[idx] = f2bf(src[(size_t)k * M + m]);
}

struct P9 { const float* p[9]; };
__global__ void k_embed(const int* __restrict__ x, P9 emb,
                        float* __restrict__ h, unsigned short* __restrict__ hb, int N) {
  int idx = blockIdx.x * 256 + threadIdx.x;
  if (idx >= N * 64) return;
  int i = idx >> 6, c = (idx & 63) << 2;
  float4 a = {0.f, 0.f, 0.f, 0.f};
  #pragma unroll
  for (int f = 0; f < 9; ++f) {
    int v = x[i * 9 + f];
    const float4 e = *(const float4*)(emb.p[f] + (size_t)v * 256 + c);
    a.x += e.x; a.y += e.y; a.z += e.z; a.w += e.w;
  }
  size_t o = (size_t)i * 256 + c;
  *(float4*)(h + o) = a;
  hb[o] = f2bf(a.x); hb[o + 1] = f2bf(a.y); hb[o + 2] = f2bf(a.z); hb[o + 3] = f2bf(a.w);
}

__global__ void k_zzinit(const float* __restrict__ h, const float* __restrict__ eps,
                         float* __restrict__ zz, int n) {
  int i = blockIdx.x * 256 + threadIdx.x; if (i >= n) return;
  zz[i] = (1.0f + eps[0]) * h[i];
}

__global__ void k_edge(const float* __restrict__ h, const int* __restrict__ ei,
                       const int* __restrict__ ea, const float* __restrict__ b0,
                       const float* __restrict__ b1, const float* __restrict__ b2,
                       float* __restrict__ zz, int E) {
  int idx = blockIdx.x * 256 + threadIdx.x;
  if (idx >= E * 64) return;
  int e = idx >> 6, c = (idx & 63) << 2;
  int src = ei[e], dst = ei[E + e];
  int a0 = ea[e * 3], a1 = ea[e * 3 + 1], a2 = ea[e * 3 + 2];
  const float4 hv = *(const float4*)(h  + (size_t)src * 256 + c);
  const float4 e0 = *(const float4*)(b0 + (size_t)a0 * 256 + c);
  const float4 e1 = *(const float4*)(b1 + (size_t)a1 * 256 + c);
  const float4 e2 = *(const float4*)(b2 + (size_t)a2 * 256 + c);
  float* zp = zz + (size_t)dst * 256 + c;
  atomicAdd(zp + 0, fmaxf(hv.x + e0.x + e1.x + e2.x, 0.f));
  atomicAdd(zp + 1, fmaxf(hv.y + e0.y + e1.y + e2.y, 0.f));
  atomicAdd(zp + 2, fmaxf(hv.z + e0.z + e1.z + e2.z, 0.f));
  atomicAdd(zp + 3, fmaxf(hv.w + e0.w + e1.w + e2.w, 0.f));
}

__global__ void k_tobf16(const float* __restrict__ s, unsigned short* __restrict__ d, int n) {
  int i = blockIdx.x * 256 + threadIdx.x; if (i >= n) return;
  d[i] = f2bf(s[i]);
}

__global__ void k_snr(const float* __restrict__ t, const float* __restrict__ s,
                      const float* __restrict__ o, unsigned short* __restrict__ d,
                      int n, int cm) {
  int i = blockIdx.x * 256 + threadIdx.x; if (i >= n) return;
  int c = i & cm;
  d[i] = f2bf(fmaxf(t[i] * s[c] + o[c], 0.f));
}

__global__ void k_colpart(const float* __restrict__ t, float* __restrict__ ps,
                          float* __restrict__ pq, int C) {
  int col = blockIdx.x * 256 + threadIdx.x;
  size_t base = (size_t)(blockIdx.y * 256) * C + col;
  float s = 0.f, q = 0.f;
  for (int r = 0; r < 256; ++r) { float v = t[base + (size_t)r * C]; s += v; q += v * v; }
  ps[blockIdx.y * C + col] = s;
  pq[blockIdx.y * C + col] = q;
}

__global__ void k_statfin(const float* __restrict__ ps, const float* __restrict__ pq,
                          const float* __restrict__ g, const float* __restrict__ b,
                          float* __restrict__ s, float* __restrict__ o, int C) {
  int c = blockIdx.x * 256 + threadIdx.x; if (c >= C) return;
  float su = 0.f, q = 0.f;
  for (int i = 0; i < 256; ++i) { su += ps[i * C + c]; q += pq[i * C + c]; }
  const float invN = 1.0f / 65536.0f;
  float m = su * invN, var = q * invN - m * m;
  float sc = g[c] * rsqrtf(var + 1e-5f);
  s[c] = sc; o[c] = b[c] - m * sc;
}

__global__ void k_combine(const float* __restrict__ r1, float* __restrict__ r2,
                          const float* __restrict__ s1, const float* __restrict__ o1,
                          const float* __restrict__ s2, const float* __restrict__ o2,
                          unsigned short* __restrict__ ob, int n) {
  int i = blockIdx.x * 256 + threadIdx.x; if (i >= n) return;
  int c = i & 255;
  float v = r1[i] * s1[c] + o1[c] + r2[i] * s2[c] + o2[c];
  r2[i] = v;           // r2 becomes "out" (f32), in-place
  ob[i] = f2bf(v);
}

__global__ void k_hupd(const float* __restrict__ r3, const float* __restrict__ s,
                       const float* __restrict__ o, float* __restrict__ h,
                       unsigned short* __restrict__ hb, int n) {
  int i = blockIdx.x * 256 + threadIdx.x; if (i >= n) return;
  int c = i & 255;
  float v = r3[i] * s[c] + o[c];
  h[i] = v; hb[i] = f2bf(v);
}

__global__ void k_pool(const float* __restrict__ h, float* __restrict__ g) {
  int gi = blockIdx.x, c = threadIdx.x;
  const float* p = h + (size_t)gi * 128 * 256 + c;
  float s = 0.f;
  for (int r = 0; r < 128; ++r) s += p[r * 256];
  g[gi * 256 + c] = s;
}

__global__ __launch_bounds__(256) void k_head(const float* __restrict__ g,
    const float* __restrict__ W1, const float* __restrict__ b1,
    const float* __restrict__ W2, const float* __restrict__ b2,
    float* __restrict__ out) {
  __shared__ float gv[256], hid[256], red[256];
  int gi = blockIdx.x, c = threadIdx.x;
  gv[c] = g[gi * 256 + c];
  __syncthreads();
  float a = b1[c];
  for (int k = 0; k < 256; ++k) a += gv[k] * W1[k * 256 + c];
  hid[c] = fmaxf(a, 0.f);
  __syncthreads();
  float o = b2[c];
  for (int k = 0; k < 256; ++k) o += hid[k] * W2[k * 256 + c];
  red[c] = o * o;
  __syncthreads();
  for (int st = 128; st > 0; st >>= 1) { if (c < st) red[c] += red[c + st]; __syncthreads(); }
  float nrm = fmaxf(sqrtf(red[0]), 1e-12f);
  out[gi * 256 + c] = o / nrm;
}

// ---------------------------------------------------------------------------
extern "C" void kernel_launch(void* const* d_in, const int* in_sizes, int n_in,
                              void* d_out, int out_size, void* d_ws, size_t ws_size,
                              hipStream_t stream) {
  (void)in_sizes; (void)n_in; (void)out_size; (void)ws_size;
  const int N = 65536, G = 512, E = 262144;
  const int* x  = (const int*)d_in[0];
  const int* ea = (const int*)d_in[1];
  const int* ei = (const int*)d_in[2];
  P9 emb; for (int i = 0; i < 9; ++i) emb.p[i] = (const float*)d_in[4 + i];
  const float* bond[3]; for (int i = 0; i < 3; ++i) bond[i] = (const float*)d_in[13 + i];
  const int LB = 16, LS = 21;
  auto L = [&](int l, int k) -> const float* { return (const float*)d_in[LB + l * LS + k]; };
  const float* pW1 = (const float*)d_in[LB + 4 * LS + 0];
  const float* pb1 = (const float*)d_in[LB + 4 * LS + 1];
  const float* pW2 = (const float*)d_in[LB + 4 * LS + 2];
  const float* pb2 = (const float*)d_in[LB + 4 * LS + 3];

  char* ws = (char*)d_ws; size_t off = 0;
  auto alloc = [&](size_t bytes) -> char* {
    char* p = ws + off; off += (bytes + 255) & ~(size_t)255; return p;
  };
  unsigned short* wt = (unsigned short*)alloc((size_t)4 * 786432 * 2); // bf16 Wt per layer
  float* h           = (float*)alloc((size_t)N * 256 * 4);
  unsigned short* hb = (unsigned short*)alloc((size_t)N * 256 * 2);
  unsigned short* bb = (unsigned short*)alloc((size_t)N * 256 * 2);    // zzb / attb / outb
  float* big         = (float*)alloc((size_t)N * 512 * 4);             // t, then r2/outf
  unsigned short* tb = (unsigned short*)alloc((size_t)N * 512 * 2);    // MLP/FFN hidden bf16
  float* r1          = (float*)alloc((size_t)N * 256 * 4);             // zz / r1 / r3
  float* ps          = (float*)alloc((size_t)256 * 512 * 4);
  float* pq          = (float*)alloc((size_t)256 * 512 * 4);
  float* sv          = (float*)alloc((size_t)8 * 512 * 4);
  float* gp          = (float*)alloc((size_t)G * 256 * 4);
  float *st = sv, *ot = sv + 512, *s1 = sv + 1024, *o1 = sv + 1536;
  float *s2 = sv + 2048, *o2 = sv + 2560, *s3 = sv + 3072, *o3 = sv + 3584;

  dim3 b256(256);
  // --- pre-transpose + bf16-convert all layer weights ---
  for (int l = 0; l < 4; ++l) {
    unsigned short* wl = wt + (size_t)l * 786432;
    k_transpose<<<512, b256, 0, stream>>>(L(l, 1),  wl + 0,      256, 512); // W1t  [512][256]
    k_transpose<<<512, b256, 0, stream>>>(L(l, 5),  wl + 131072, 512, 256); // W2t  [256][512]
    k_transpose<<<768, b256, 0, stream>>>(L(l, 7),  wl + 262144, 256, 768); // Wqkvt[768][256]
    k_transpose<<<256, b256, 0, stream>>>(L(l, 9),  wl + 458752, 256, 256); // Wot
    k_transpose<<<512, b256, 0, stream>>>(L(l, 17), wl + 524288, 256, 512); // Wm1t
    k_transpose<<<512, b256, 0, stream>>>(L(l, 19), wl + 655360, 512, 256); // Wm2t
  }
  k_embed<<<(N * 64) / 256, b256, 0, stream>>>(x, emb, h, hb, N);

  for (int l = 0; l < 4; ++l) {
    unsigned short* wl = wt + (size_t)l * 786432;
    float* t  = big;   // [N,512]
    float* r2 = big;   // [N,256] reuses big after t is dead
    // --- GINEConv local branch ---
    k_zzinit<<<(N * 256) / 256, b256, 0, stream>>>(h, L(l, 0), r1, N * 256);
    k_edge<<<(E * 64) / 256, b256, 0, stream>>>(h, ei, ea, bond[0], bond[1], bond[2], r1, E);
    k_tobf16<<<(N * 256) / 256, b256, 0, stream>>>(r1, bb, N * 256);
    k_gemm<0><<<dim3(512, 4), b256, 0, stream>>>(bb, wl + 0, L(l, 2), nullptr, t, nullptr, 256, 512);
    k_colpart<<<dim3(2, 256), b256, 0, stream>>>(t, ps, pq, 512);
    k_statfin<<<2, b256, 0, stream>>>(ps, pq, L(l, 3), L(l, 4), st, ot, 512);
    k_snr<<<(N * 512) / 256, b256, 0, stream>>>(t, st, ot, tb, N * 512, 511);
    k_gemm<1><<<dim3(512, 2), b256, 0, stream>>>(tb, wl + 131072, L(l, 6), h, r1, nullptr, 512, 256);
    k_colpart<<<dim3(1, 256), b256, 0, stream>>>(r1, ps, pq, 256);
    k_statfin<<<1, b256, 0, stream>>>(ps, pq, L(l, 11), L(l, 12), s1, o1, 256);
    // --- global branch: per-graph MHA, then Wo + residual ---
    k_attn<<<512, b256, 0, stream>>>(hb, wl + 262144, L(l, 8), bb);
    k_gemm<1><<<dim3(512, 2), b256, 0, stream>>>(bb, wl + 458752, L(l, 10), h, r2, nullptr, 256, 256);
    k_colpart<<<dim3(1, 256), b256, 0, stream>>>(r2, ps, pq, 256);
    k_statfin<<<1, b256, 0, stream>>>(ps, pq, L(l, 13), L(l, 14), s2, o2, 256);
    // --- combine + FFN ---
    k_combine<<<(N * 256) / 256, b256, 0, stream>>>(r1, r2, s1, o1, s2, o2, bb, N * 256);
    k_gemm<2><<<dim3(512, 4), b256, 0, stream>>>(bb, wl + 524288, L(l, 18), nullptr, nullptr, tb, 256, 512);
    k_gemm<1><<<dim3(512, 2), b256, 0, stream>>>(tb, wl + 655360, L(l, 20), r2, r1, nullptr, 512, 256);
    k_colpart<<<dim3(1, 256), b256, 0, stream>>>(r1, ps, pq, 256);
    k_statfin<<<1, b256, 0, stream>>>(ps, pq, L(l, 15), L(l, 16), s3, o3, 256);
    k_hupd<<<(N * 256) / 256, b256, 0, stream>>>(r1, s3, o3, h, hb, N * 256);
  }
  k_pool<<<G, b256, 0, stream>>>(h, gp);
  k_head<<<G, b256, 0, stream>>>(gp, pW1, pb1, pW2, pb2, (float*)d_out);
}
